// arxiv_classifier_14834817040536
// MI455X (gfx1250) — compile-verified, matrix operand order checked by reading, not run on hardware
//
#include <hip/hip_runtime.h>
#include <hip/hip_bf16.h>

// ---------------------------------------------------------------------------
// GAT(3 layers) + JK-BiLSTM + attention + MLP classifier for MI455X (gfx1250)
// Dense GEMMs run through v_wmma_f32_16x16x32_f16 with double-buffered
// LDS-staged tiles; each wave computes a 16x32 output (2 WMMAs / K-step).
// ---------------------------------------------------------------------------

typedef __attribute__((ext_vector_type(16))) _Float16 v16h;
typedef __attribute__((ext_vector_type(8)))  _Float16 v8h;
typedef __attribute__((ext_vector_type(4)))  _Float16 v4h;
typedef __attribute__((ext_vector_type(8)))  float    v8f;

#define NEGS 0.01f
#define EPSB 1e-5f

// GEMM epilogue / option flags
#define EPI_BIAS   1
#define EPI_LEAKY  2
#define EPI_ACC    4
#define OPT_TRANSB 8

#define LDSTR 40               // LDS row stride in halves (80B)
#define ASZ   (64 * LDSTR)     // one A buffer: 64 rows x 32 k
#define BSZ   (64 * LDSTR)     // one B buffer: 64 n  x 32 k

__device__ __forceinline__ float leakyf(float v, float s) { return v > 0.f ? v : s * v; }
__device__ __forceinline__ float sigmf(float v) { return 1.f / (1.f + __expf(-v)); }
__device__ __forceinline__ int   clampi(int v, int hi) { return v < hi ? v : hi; }

// ---------------------------------------------------------------------------
// WMMA GEMM: C[M,N] = A[M,K] * B (+C if ACC) (+bias) (leaky)
// A row-major f32 (lda). B row-major f32 (ldb), or B[n,k] when OPT_TRANSB.
// CONTRACT: K % 32 == 0; lda/ldb multiples of 4; base pointers 16B-aligned.
// Block: 256 threads = 8 waves; block tile 64x64; wave tile 16x32 (2 acc).
// Double-buffered LDS: one barrier per K-step; staging of tile k+1 overlaps
// the two WMMAs of tile k. OOB rows/cols CLAMPED during staging (never
// stored), keeping the K-loop free of per-lane branches.
// ---------------------------------------------------------------------------
__global__ void __launch_bounds__(256)
wmma_gemm_kernel(const float* __restrict__ A, const float* __restrict__ B,
                 float* __restrict__ C, int M, int N, int K,
                 int lda, int ldb, int ldc,
                 const float* __restrict__ bias, float leak, int flags)
{
    __shared__ __align__(16) _Float16 sA[2 * ASZ];
    __shared__ __align__(16) _Float16 sB[2 * BSZ];

    const int tid  = threadIdx.x;
    const int lane = tid & 31;
    const int wave = tid >> 5;                         // 0..7
    const int rowBase = blockIdx.y * 64;
    const int colBase = blockIdx.x * 64;
    const int mrow = lane & 15;
    const int hi   = lane >> 4;
    const int trow   = rowBase + (wave & 3) * 16;      // wave C rows
    const int ccBase = colBase + (wave >> 2) * 32 + mrow;

    // block-uniform full-tile predicate (scalar branch only)
    const bool fullN = (colBase + 64 <= N);

    // staging coordinates (loop-invariant)
    const int srow = tid >> 3;                         // 0..31 (+32 second chunk)
    const int sc4  = (tid & 7) << 2;                   // 0,4,..,28
    const int agr0 = clampi(rowBase + srow,      M - 1);
    const int agr1 = clampi(rowBase + srow + 32, M - 1);
    const int tbn0 = clampi(colBase + srow,      N - 1);  // transB rows
    const int tbn1 = clampi(colBase + srow + 32, N - 1);

    auto stageA = [&](int buf, int k0) {
        const float4 v0 = *(const float4*)(A + (size_t)agr0 * lda + k0 + sc4);
        const float4 v1 = *(const float4*)(A + (size_t)agr1 * lda + k0 + sc4);
        v4h h0; h0[0] = (_Float16)v0.x; h0[1] = (_Float16)v0.y;
                h0[2] = (_Float16)v0.z; h0[3] = (_Float16)v0.w;
        v4h h1; h1[0] = (_Float16)v1.x; h1[1] = (_Float16)v1.y;
                h1[2] = (_Float16)v1.z; h1[3] = (_Float16)v1.w;
        _Float16* dp = sA + buf * ASZ;
        *(v4h*)(dp + srow * LDSTR + sc4)        = h0;
        *(v4h*)(dp + (srow + 32) * LDSTR + sc4) = h1;
        if (k0 + 64 < K) {
            __builtin_prefetch(A + (size_t)agr0 * lda + k0 + 64 + sc4, 0, 0);
            __builtin_prefetch(A + (size_t)agr1 * lda + k0 + 64 + sc4, 0, 0);
        }
    };

    auto stageB = [&](int buf, int k0) {
        _Float16* dp = sB + buf * BSZ;
        if (flags & OPT_TRANSB) {
            // B[n,k]: direct copy, rows clamped
            const float4 v0 = *(const float4*)(B + (size_t)tbn0 * ldb + k0 + sc4);
            const float4 v1 = *(const float4*)(B + (size_t)tbn1 * ldb + k0 + sc4);
            v4h h0; h0[0] = (_Float16)v0.x; h0[1] = (_Float16)v0.y;
                    h0[2] = (_Float16)v0.z; h0[3] = (_Float16)v0.w;
            v4h h1; h1[0] = (_Float16)v1.x; h1[1] = (_Float16)v1.y;
                    h1[2] = (_Float16)v1.z; h1[3] = (_Float16)v1.w;
            *(v4h*)(dp + srow * LDSTR + sc4)        = h0;
            *(v4h*)(dp + (srow + 32) * LDSTR + sc4) = h1;
            if (k0 + 64 < K) {
                __builtin_prefetch(B + (size_t)tbn0 * ldb + k0 + 64 + sc4, 0, 0);
                __builtin_prefetch(B + (size_t)tbn1 * ldb + k0 + 64 + sc4, 0, 0);
            }
        } else {
            // B[k,n]: transpose into LDS; kl = k row, two 4-col groups
            const int kl = srow;                       // 0..31
            const int gk = k0 + kl;
            if (fullN) {                               // block-uniform branch
#pragma unroll
                for (int gsel = 0; gsel < 2; ++gsel) {
                    const int n4 = sc4 + gsel * 32;
                    const float4 vb = *(const float4*)(B + (size_t)gk * ldb + colBase + n4);
                    dp[(n4 + 0) * LDSTR + kl] = (_Float16)vb.x;
                    dp[(n4 + 1) * LDSTR + kl] = (_Float16)vb.y;
                    dp[(n4 + 2) * LDSTR + kl] = (_Float16)vb.z;
                    dp[(n4 + 3) * LDSTR + kl] = (_Float16)vb.w;
                }
            } else {
#pragma unroll
                for (int gsel = 0; gsel < 2; ++gsel) {
                    const int n4 = sc4 + gsel * 32;
#pragma unroll
                    for (int j = 0; j < 4; ++j) {
                        const int cn = clampi(colBase + n4 + j, N - 1);
                        dp[(n4 + j) * LDSTR + kl] = (_Float16)B[(size_t)gk * ldb + cn];
                    }
                }
            }
        }
    };

    v8f acc[2] = {{}, {}};
    if (flags & EPI_ACC) {
#pragma unroll
        for (int hf2 = 0; hf2 < 2; ++hf2) {
            const int cc = clampi(ccBase + hf2 * 16, N - 1);
#pragma unroll
            for (int r = 0; r < 8; ++r) {
                const int rr = clampi(trow + r + 8 * hi, M - 1);
                acc[hf2][r] = C[(size_t)rr * ldc + cc];
            }
        }
    }

    // fragment read offsets (ISA 16-bit 16x32 layout: lanes 0-15 supply
    // K {0..7,16..23}; lanes 16-31 supply K {8..15,24..31})
    const int arow = (wave & 3) * 16 + mrow;
    const int bn0  = (wave >> 2) * 32 + mrow;

    // software pipeline: stage tile 0, then 1 barrier per K-step
    stageA(0, 0);
    stageB(0, 0);
    int cur = 0;
    for (int k0 = 0; k0 < K; k0 += 32) {
        __syncthreads();
        const _Float16* pA = sA + cur * ASZ;
        const _Float16* pB = sB + cur * BSZ;
        const v8h a0 = *(const v8h*)(pA + arow * LDSTR + hi * 8);
        const v8h a1 = *(const v8h*)(pA + arow * LDSTR + 16 + hi * 8);
        const v8h b00 = *(const v8h*)(pB + bn0 * LDSTR + hi * 8);
        const v8h b01 = *(const v8h*)(pB + bn0 * LDSTR + 16 + hi * 8);
        const v8h b10 = *(const v8h*)(pB + (bn0 + 16) * LDSTR + hi * 8);
        const v8h b11 = *(const v8h*)(pB + (bn0 + 16) * LDSTR + 16 + hi * 8);
        const v16h av  = __builtin_shufflevector(a0, a1,
            0,1,2,3,4,5,6,7,8,9,10,11,12,13,14,15);
        const v16h bv0 = __builtin_shufflevector(b00, b01,
            0,1,2,3,4,5,6,7,8,9,10,11,12,13,14,15);
        const v16h bv1 = __builtin_shufflevector(b10, b11,
            0,1,2,3,4,5,6,7,8,9,10,11,12,13,14,15);
        if (k0 + 32 < K) {                 // stage next tile (overlaps WMMAs)
            stageA(cur ^ 1, k0 + 32);
            stageB(cur ^ 1, k0 + 32);
        }
        acc[0] = __builtin_amdgcn_wmma_f32_16x16x32_f16(
                     false, av, false, bv0, (short)0, acc[0], false, false);
        acc[1] = __builtin_amdgcn_wmma_f32_16x16x32_f16(
                     false, av, false, bv1, (short)0, acc[1], false, false);
        cur ^= 1;
    }

#pragma unroll
    for (int hf2 = 0; hf2 < 2; ++hf2) {
        const int cc = ccBase + hf2 * 16;
        if (cc < N) {
            const float badd = (flags & EPI_BIAS) ? bias[cc] : 0.f;
#pragma unroll
            for (int r = 0; r < 8; ++r) {
                const int rr = trow + r + 8 * hi;
                if (rr < M) {
                    float v = acc[hf2][r] + badd;
                    if (flags & EPI_LEAKY) v = leakyf(v, leak);
                    C[(size_t)rr * ldc + cc] = v;
                }
            }
        }
    }
}

// ---------------------------------------------------------------------------
// Utility fills
// ---------------------------------------------------------------------------
__global__ void fill_f32_kernel(float* p, float v, int n) {
    int i = blockIdx.x * blockDim.x + threadIdx.x;
    if (i < n) p[i] = v;
}
__global__ void fill_u32_kernel(unsigned* p, unsigned v, int n) {
    int i = blockIdx.x * blockDim.x + threadIdx.x;
    if (i < n) p[i] = v;
}
__global__ void add_vec_kernel(const float* a, const float* b, float* o, int n) {
    int i = blockIdx.x * blockDim.x + threadIdx.x;
    if (i < n) o[i] = a[i] + b[i];
}
__global__ void copy_x_to_xs_kernel(const float* __restrict__ x, float* __restrict__ xs, int N) {
    int i = blockIdx.x * blockDim.x + threadIdx.x;        // N*128
    if (i >= N * 128) return;
    int n = i >> 7, c = i & 127;
    xs[(size_t)n * 512 + c] = x[(size_t)n * 128 + c];
}

// ---------------------------------------------------------------------------
// GAT: per-node attention coefficients a_s/a_d (warp per node-head)
// ---------------------------------------------------------------------------
__global__ void gat_coef_kernel(const float* __restrict__ hf,
                                const float* __restrict__ a_src,
                                const float* __restrict__ a_dst,
                                float* __restrict__ as_, float* __restrict__ ad_, int N)
{
    int warp = (blockIdx.x * blockDim.x + threadIdx.x) >> 5;
    int lane = threadIdx.x & 31;
    if (warp >= N * 2) return;
    int n = warp >> 1, hd = warp & 1;
    const float* row = hf + (size_t)n * 256 + hd * 128;
    const float* ws  = a_src + hd * 128;
    const float* wd  = a_dst + hd * 128;
    float s = 0.f, d = 0.f;
    for (int i = lane; i < 128; i += 32) {
        float hv = row[i];
        s += hv * ws[i];
        d += hv * wd[i];
    }
#pragma unroll
    for (int off = 16; off > 0; off >>= 1) {
        s += __shfl_down(s, off, 32);
        d += __shfl_down(d, off, 32);
    }
    if (lane == 0) { as_[n * 2 + hd] = s; ad_[n * 2 + hd] = d; }
}

// ordered-uint float encoding for atomic max
__device__ __forceinline__ unsigned f2ord(float v) {
    unsigned u = __float_as_uint(v);
    return (u & 0x80000000u) ? ~u : (u | 0x80000000u);
}
__device__ __forceinline__ float ord2f(unsigned u) {
    return (u & 0x80000000u) ? __uint_as_float(u ^ 0x80000000u) : __uint_as_float(~u);
}

__global__ void edge_max_kernel(const int* __restrict__ src, const int* __restrict__ dst,
                                const float* __restrict__ as_, const float* __restrict__ ad_,
                                unsigned* __restrict__ menc, int E)
{
    int i = blockIdx.x * blockDim.x + threadIdx.x;        // E*2
    if (i >= E * 2) return;
    int e = i >> 1, hd = i & 1;
    int sN = src[e], dN = dst[e];
    float v = leakyf(as_[sN * 2 + hd] + ad_[dN * 2 + hd], 0.2f);
    atomicMax(&menc[dN * 2 + hd], f2ord(v));
}

__global__ void edge_exp_kernel(const int* __restrict__ src, const int* __restrict__ dst,
                                const float* __restrict__ as_, const float* __restrict__ ad_,
                                const unsigned* __restrict__ menc,
                                float* __restrict__ eexp, float* __restrict__ ssum, int E)
{
    int i = blockIdx.x * blockDim.x + threadIdx.x;        // E*2
    if (i >= E * 2) return;
    int e = i >> 1, hd = i & 1;
    int sN = src[e], dN = dst[e];
    float v  = leakyf(as_[sN * 2 + hd] + ad_[dN * 2 + hd], 0.2f);
    float m  = ord2f(menc[dN * 2 + hd]);
    float ex = __expf(v - m);
    eexp[(size_t)e * 2 + hd] = ex;
    atomicAdd(&ssum[dN * 2 + hd], ex);
}

// per-edge message scatter: block = one edge, 256 threads = 2 heads x 128 dims
__global__ void edge_msg_kernel(const int* __restrict__ src, const int* __restrict__ dst,
                                const float* __restrict__ eexp, const float* __restrict__ ssum,
                                const float* __restrict__ hf, float* __restrict__ outacc, int E)
{
    int e = blockIdx.x;
    int t = threadIdx.x, hd = t >> 7, d = t & 127;
    int sN = src[e], dN = dst[e];
    float alpha = eexp[(size_t)e * 2 + hd] / (ssum[dN * 2 + hd] + 1e-16f);
    atomicAdd(&outacc[(size_t)dN * 256 + hd * 128 + d],
              hf[(size_t)sN * 256 + hd * 128 + d] * alpha);
}

// mean over heads + layer bias + residual -> v_pre (first N1 rows)
__global__ void gat_post_kernel(const float* __restrict__ outacc, const float* __restrict__ prevx,
                                const float* __restrict__ bias, float* __restrict__ vpre, int rows)
{
    int i = blockIdx.x * blockDim.x + threadIdx.x;        // rows*128
    if (i >= rows * 128) return;
    int r = i >> 7, c = i & 127;
    float v = 0.5f * (outacc[(size_t)r * 256 + c] + outacc[(size_t)r * 256 + 128 + c]);
    vpre[i] = v + bias[c] + prevx[(size_t)r * 128 + c];
}

// ---------------------------------------------------------------------------
// BatchNorm (population stats) over axis 0; block per column
// ---------------------------------------------------------------------------
__global__ void bn_stats_kernel(const float* __restrict__ v, int rows, int cols,
                                float* __restrict__ mu, float* __restrict__ rsig)
{
    int c = blockIdx.x;
    float sum = 0.f, sq = 0.f;
    for (int r = threadIdx.x; r < rows; r += blockDim.x) {
        float x = v[(size_t)r * cols + c];
        sum += x; sq += x * x;
    }
    __shared__ float ss[256], sq2[256];
    ss[threadIdx.x] = sum; sq2[threadIdx.x] = sq;
    __syncthreads();
    for (int s = 128; s > 0; s >>= 1) {
        if (threadIdx.x < s) { ss[threadIdx.x] += ss[threadIdx.x + s]; sq2[threadIdx.x] += sq2[threadIdx.x + s]; }
        __syncthreads();
    }
    if (threadIdx.x == 0) {
        float m   = ss[0] / rows;
        float var = sq2[0] / rows - m * m;
        mu[c] = m; rsig[c] = rsqrtf(var + EPSB);
    }
}

// GAT: BN + leaky, write next-layer x and the JK slice into xs
__global__ void gat_bn_act_kernel(const float* __restrict__ vpre, int rows,
                                  const float* __restrict__ mu, const float* __restrict__ rsig,
                                  const float* __restrict__ gamma, const float* __restrict__ beta,
                                  float* __restrict__ xnext, float* __restrict__ xs, int slot)
{
    int i = blockIdx.x * blockDim.x + threadIdx.x;        // rows*128
    if (i >= rows * 128) return;
    int r = i >> 7, c = i & 127;
    float v = (vpre[i] - mu[c]) * rsig[c] * gamma[c] + beta[c];
    v = leakyf(v, NEGS);
    xnext[i] = v;
    if (r < 20000) xs[(size_t)r * 512 + slot * 128 + c] = v;
}

// BN apply in-place (FC path; no activation)
__global__ void bn_apply_kernel(float* __restrict__ v, int rows, int cols,
                                const float* __restrict__ mu, const float* __restrict__ rsig,
                                const float* __restrict__ gamma, const float* __restrict__ beta)
{
    int i = blockIdx.x * blockDim.x + threadIdx.x;
    if (i >= rows * cols) return;
    int c = i % cols;
    v[i] = (v[i] - mu[c]) * rsig[c] * gamma[c] + beta[c];
}

// ---------------------------------------------------------------------------
// LSTM gate nonlinearity: z[N,1024] -> h,c[N,256]
// ---------------------------------------------------------------------------
__global__ void lstm_gates_kernel(const float* __restrict__ z,
                                  float* __restrict__ h, float* __restrict__ c, int N)
{
    int i = blockIdx.x * blockDim.x + threadIdx.x;        // N*256
    if (i >= N * 256) return;
    int n = i >> 8, j = i & 255;
    const float* zr = z + (size_t)n * 1024;
    float zi = zr[j], zf = zr[256 + j], zg = zr[512 + j], zo = zr[768 + j];
    float cv = sigmf(zf) * c[i] + sigmf(zi) * tanhf(zg);
    c[i] = cv;
    h[i] = sigmf(zo) * tanhf(cv);
}

// accumulate attention score: score[n,t] += dot(h[n,:], attW_half)
__global__ void score_accum_kernel(const float* __restrict__ h, const float* __restrict__ attWh,
                                   float* __restrict__ score, int t, int N)
{
    int warp = (blockIdx.x * blockDim.x + threadIdx.x) >> 5;
    int lane = threadIdx.x & 31;
    if (warp >= N) return;
    float s = 0.f;
    for (int i = lane; i < 256; i += 32) s += h[(size_t)warp * 256 + i] * attWh[i];
#pragma unroll
    for (int off = 16; off > 0; off >>= 1) s += __shfl_down(s, off, 32);
    if (lane == 0) score[warp * 4 + t] += s;
}

// softmax over 4 steps + weighted sum of xs -> g[N,128]
__global__ void jk_gather_kernel(const float* __restrict__ score, const float* __restrict__ xs,
                                 float* __restrict__ g, int N)
{
    int i = blockIdx.x * blockDim.x + threadIdx.x;        // N*128
    if (i >= N * 128) return;
    int n = i >> 7, c = i & 127;
    float s0 = score[n * 4 + 0], s1 = score[n * 4 + 1];
    float s2 = score[n * 4 + 2], s3 = score[n * 4 + 3];
    float m = fmaxf(fmaxf(s0, s1), fmaxf(s2, s3));
    float e0 = __expf(s0 - m), e1 = __expf(s1 - m), e2 = __expf(s2 - m), e3 = __expf(s3 - m);
    float inv = 1.f / (e0 + e1 + e2 + e3);
    const float* xr = xs + (size_t)n * 512 + c;
    g[i] = (e0 * xr[0] + e1 * xr[128] + e2 * xr[256] + e3 * xr[384]) * inv;
}

// ---------------------------------------------------------------------------
// Host orchestration
// ---------------------------------------------------------------------------
static inline int cdiv(int a, int b) { return (a + b - 1) / b; }

extern "C" void kernel_launch(void* const* d_in, const int* in_sizes, int n_in,
                              void* d_out, int out_size, void* d_ws, size_t ws_size,
                              hipStream_t stream)
{
    (void)in_sizes; (void)n_in; (void)out_size; (void)ws_size;
    const int NSv[4] = {50000, 40000, 30000, 20000};
    const int E = 600000, NL = 20000;

    // Input layout: recursive insertion-order flatten of setup_inputs() dict:
    //  0: x
    //  1+6k..6+6k (k=0..2): gat[k] {W, a_src, a_dst, bias, gamma, beta}
    //  19..28: jk {Wih_f, Whh_f, bih_f, bhh_f, Wih_b, Whh_b, bih_b, bhh_b, attW, attb}
    //  29..34: fcn {W1, b1, gamma, beta, W2, b2}
    //  35..37: edge_index0..2 (int32, shape (2,E))
    const float* X = (const float*)d_in[0];
    const float *gW[3], *gAs[3], *gAd[3], *gB[3], *gG[3], *gBe[3];
    for (int k = 0; k < 3; ++k) {
        gW[k]  = (const float*)d_in[1 + 6 * k];
        gAs[k] = (const float*)d_in[2 + 6 * k];
        gAd[k] = (const float*)d_in[3 + 6 * k];
        gB[k]  = (const float*)d_in[4 + 6 * k];
        gG[k]  = (const float*)d_in[5 + 6 * k];
        gBe[k] = (const float*)d_in[6 + 6 * k];
    }
    const float* Wih_f = (const float*)d_in[19];
    const float* Whh_f = (const float*)d_in[20];
    const float* bih_f = (const float*)d_in[21];
    const float* bhh_f = (const float*)d_in[22];
    const float* Wih_b = (const float*)d_in[23];
    const float* Whh_b = (const float*)d_in[24];
    const float* bih_b = (const float*)d_in[25];
    const float* bhh_b = (const float*)d_in[26];
    const float* attW  = (const float*)d_in[27];
    const float* W1 = (const float*)d_in[29];
    const float* b1 = (const float*)d_in[30];
    const float* fG = (const float*)d_in[31];
    const float* fB = (const float*)d_in[32];
    const float* W2 = (const float*)d_in[33];
    const float* b2 = (const float*)d_in[34];
    const int* eidx[3] = {(const int*)d_in[35], (const int*)d_in[36], (const int*)d_in[37]};

    // ---- workspace carving (persistent + phase-aliased union) ----
    char* base = (char*)d_ws;
    size_t off = 0;
    auto WALLOC = [&](size_t bytes) -> void* {
        void* p = base + off;
        off = (off + bytes + 255) & ~(size_t)255;
        return p;
    };
    float* XS    = (float*)WALLOC((size_t)NL * 512 * 4);   // (20000,4,128)
    float* SCORE = (float*)WALLOC((size_t)NL * 4 * 4);
    size_t mark = off;

    // GAT phase scratch
    float*    B0   = (float*)WALLOC((size_t)50000 * 128 * 4);
    float*    B1   = (float*)WALLOC((size_t)50000 * 128 * 4);
    float*    Hf   = (float*)WALLOC((size_t)50000 * 256 * 4);
    float*    Oa   = (float*)WALLOC((size_t)50000 * 256 * 4);
    float*    VP   = (float*)WALLOC((size_t)40000 * 128 * 4);
    float*    AS   = (float*)WALLOC((size_t)50000 * 2 * 4);
    float*    AD   = (float*)WALLOC((size_t)50000 * 2 * 4);
    unsigned* MEnc = (unsigned*)WALLOC((size_t)50000 * 2 * 4);
    float*    SS   = (float*)WALLOC((size_t)50000 * 2 * 4);
    float*    EEXP = (float*)WALLOC((size_t)E * 2 * 4);
    float*    MUg  = (float*)WALLOC(256 * 4);
    float*    RSg  = (float*)WALLOC(256 * 4);

    // JK slice 0 = x[:20000]
    copy_x_to_xs_kernel<<<cdiv(NL * 128, 256), 256, 0, stream>>>(X, XS, NL);

    // ---- GAT layers ----
    const float* xcur = X;
    for (int k = 0; k < 3; ++k) {
        const int Nk = NSv[k], Nk1 = NSv[k + 1];
        const int* src = eidx[k];
        const int* dst = eidx[k] + E;

        // h = x @ W  (Nk x 256)
        wmma_gemm_kernel<<<dim3(cdiv(256, 64), cdiv(Nk, 64)), 256, 0, stream>>>(
            xcur, gW[k], Hf, Nk, 256, 128, 128, 256, 256, nullptr, 0.f, 0);

        gat_coef_kernel<<<cdiv(Nk * 2, 8), 256, 0, stream>>>(Hf, gAs[k], gAd[k], AS, AD, Nk);

        // ordered-uint encoding of -inf == ~bits(-inf) = 0x007FFFFF
        fill_u32_kernel<<<cdiv(Nk1 * 2, 256), 256, 0, stream>>>(MEnc, 0x007FFFFFu, Nk1 * 2);
        fill_f32_kernel<<<cdiv(Nk1 * 2, 256), 256, 0, stream>>>(SS, 0.f, Nk1 * 2);
        fill_f32_kernel<<<cdiv(Nk1 * 256, 256), 256, 0, stream>>>(Oa, 0.f, Nk1 * 256);

        edge_max_kernel<<<cdiv(E * 2, 256), 256, 0, stream>>>(src, dst, AS, AD, MEnc, E);
        edge_exp_kernel<<<cdiv(E * 2, 256), 256, 0, stream>>>(src, dst, AS, AD, MEnc, EEXP, SS, E);
        edge_msg_kernel<<<E, 256, 0, stream>>>(src, dst, EEXP, SS, Hf, Oa, E);

        gat_post_kernel<<<cdiv(Nk1 * 128, 256), 256, 0, stream>>>(Oa, xcur, gB[k], VP, Nk1);
        bn_stats_kernel<<<128, 256, 0, stream>>>(VP, Nk1, 128, MUg, RSg);

        float* xnext = (k & 1) ? B1 : B0;
        gat_bn_act_kernel<<<cdiv(Nk1 * 128, 256), 256, 0, stream>>>(
            VP, Nk1, MUg, RSg, gG[k], gBe[k], xnext, XS, k + 1);
        xcur = xnext;
    }

    // ---- LSTM / FC phase (alias over GAT scratch) ----
    off = mark;
    float* ZB    = (float*)WALLOC((size_t)NL * 1024 * 4);
    float* HC    = (float*)WALLOC((size_t)NL * 256 * 4);
    float* CC    = (float*)WALLOC((size_t)NL * 256 * 4);
    float* G     = (float*)WALLOC((size_t)NL * 128 * 4);
    float* HFC   = (float*)WALLOC((size_t)NL * 256 * 4);
    float* BIASF = (float*)WALLOC(1024 * 4);
    float* BIASB = (float*)WALLOC(1024 * 4);
    float* MU2   = (float*)WALLOC(256 * 4);
    float* RS2   = (float*)WALLOC(256 * 4);

    add_vec_kernel<<<cdiv(1024, 256), 256, 0, stream>>>(bih_f, bhh_f, BIASF, 1024);
    add_vec_kernel<<<cdiv(1024, 256), 256, 0, stream>>>(bih_b, bhh_b, BIASB, 1024);
    fill_f32_kernel<<<cdiv(NL * 4, 256), 256, 0, stream>>>(SCORE, 0.f, NL * 4);

    const dim3 gLSTM(cdiv(1024, 64), cdiv(NL, 64));
    for (int dir = 0; dir < 2; ++dir) {
        const float* Wih  = dir ? Wih_b : Wih_f;
        const float* Whh  = dir ? Whh_b : Whh_f;
        const float* bias = dir ? BIASB : BIASF;
        const float* attWh = attW + dir * 256;   // hcat = [hf | hb]

        fill_f32_kernel<<<cdiv(NL * 256, 256), 256, 0, stream>>>(HC, 0.f, NL * 256);
        fill_f32_kernel<<<cdiv(NL * 256, 256), 256, 0, stream>>>(CC, 0.f, NL * 256);

        for (int s = 0; s < 4; ++s) {
            const int t = dir ? (3 - s) : s;
            // z = xs[:,t,:] @ Wih.T + (bih+bhh)
            wmma_gemm_kernel<<<gLSTM, 256, 0, stream>>>(
                XS + t * 128, Wih, ZB, NL, 1024, 128, 512, 128, 1024,
                bias, 0.f, EPI_BIAS | OPT_TRANSB);
            // z += h @ Whh.T
            wmma_gemm_kernel<<<gLSTM, 256, 0, stream>>>(
                HC, Whh, ZB, NL, 1024, 256, 256, 256, 1024,
                nullptr, 0.f, EPI_ACC | OPT_TRANSB);
            lstm_gates_kernel<<<cdiv(NL * 256, 256), 256, 0, stream>>>(ZB, HC, CC, NL);
            score_accum_kernel<<<cdiv(NL, 8), 256, 0, stream>>>(HC, attWh, SCORE, t, NL);
        }
    }

    // attention softmax over t (attb cancels) + weighted sum of xs -> g
    jk_gather_kernel<<<cdiv(NL * 128, 256), 256, 0, stream>>>(SCORE, XS, G, NL);

    // hfc = leaky(g @ W1 + b1)
    wmma_gemm_kernel<<<dim3(cdiv(256, 64), cdiv(NL, 64)), 256, 0, stream>>>(
        G, W1, HFC, NL, 256, 128, 128, 256, 256, b1, NEGS, EPI_BIAS | EPI_LEAKY);
    // BN
    bn_stats_kernel<<<256, 256, 0, stream>>>(HFC, NL, 256, MU2, RS2);
    bn_apply_kernel<<<cdiv(NL * 256, 256), 256, 0, stream>>>(HFC, NL, 256, MU2, RS2, fG, fB);
    // out = hfc @ W2 + b2  -> (20000, 40)
    wmma_gemm_kernel<<<dim3(cdiv(40, 64), cdiv(NL, 64)), 256, 0, stream>>>(
        HFC, W2, (float*)d_out, NL, 40, 256, 256, 40, 40, b2, 0.f, EPI_BIAS);
}
